// MoDTransformerStack_61615600828934
// MI455X (gfx1250) — matrix operand order repaired and checked by reading, not compiled, not run on hardware
//
#include <hip/hip_runtime.h>
#include <hip/hip_bf16.h>
#include <math.h>

// ---------------------------------------------------------------------------
// MoD transformer stack for MI455X (gfx1250, wave32, WMMA).
// GEMM-shaped math -> v_wmma_f32_16x16x32_bf16.
// Tile staging -> global_load_async_to_lds_b128 (ASYNCcnt) when available,
// double-buffered so DMA of step t+1 overlaps WMMA of step t.
// ---------------------------------------------------------------------------

typedef __bf16 bf16;
typedef __attribute__((ext_vector_type(16))) __bf16 v16bf;
typedef __attribute__((ext_vector_type(8)))  __bf16 v8bf;
typedef __attribute__((ext_vector_type(8)))  float  v8f;
typedef __attribute__((ext_vector_type(4)))  int    v4i;

#define WMMA_BF16(a, b, c) \
  __builtin_amdgcn_wmma_f32_16x16x32_bf16(false, (a), false, (b), (short)0, (c), false, false)

#ifndef __has_builtin
#define __has_builtin(x) 0
#endif

#if __has_builtin(__builtin_amdgcn_global_load_async_to_lds_b128)
#define HAVE_ASYNC_LDS 1
#else
#define HAVE_ASYNC_LDS 0
#endif

#if HAVE_ASYNC_LDS
#  if __has_builtin(__builtin_amdgcn_s_wait_asynccnt)
#    define WAIT_ASYNC() __builtin_amdgcn_s_wait_asynccnt(0)
#  else
#    define WAIT_ASYNC() asm volatile("s_wait_asynccnt 0" ::: "memory")
#  endif
#else
#  define WAIT_ASYNC() ((void)0)
#endif

// Copy 16 contiguous bytes (4 floats) global -> LDS, per lane.
// Builtin signature (from hipcc diagnostic): (v4i AS1*, v4i AS3*, Imm, Imm).
__device__ __forceinline__ void cp16(float* l, const float* g) {
#if HAVE_ASYNC_LDS
  __builtin_amdgcn_global_load_async_to_lds_b128(
      (__attribute__((address_space(1))) v4i*)(const_cast<float*>(g)),
      (__attribute__((address_space(3))) v4i*)(l), 0, 0);
#else
  *(float4*)l = *(const float4*)g;
#endif
}

static constexpr int Bc  = 4;
static constexpr int Sc  = 2048;
static constexpr int Dc  = 1024;
static constexpr int Hc  = 16;
static constexpr int HDc = 64;
static constexpr int Lc  = 6;
static constexpr int FFc = 4096;
static constexpr int DRc = 256;
static constexpr int Kc  = 1024;   // top-k per batch (= S/2)

__device__ __forceinline__ float gelu_exact(float x) {
  return 0.5f * x * (1.0f + erff(x * 0.70710678118654752f));
}

// ---------------------------------------------------------------------------
// GEMM: C[M,N] = act(A[M,Kd] @ W[Kd,N] + bias) (+ residual), template flags.
// Block tile 128(M) x 64(N), 8 waves, each wave 32x32 (2x2 WMMA subtiles),
// K-step 32, double-buffered async staging (fp32 in LDS, cvt at frag load).
// M%128==0, N%64==0, Kd%32==0 guaranteed by the launch sites.
// ---------------------------------------------------------------------------
template<int ACT, int RSD>
__global__ __launch_bounds__(256) void gemm_bf16(
    const float* __restrict__ A, const float* __restrict__ W,
    const float* __restrict__ bias, const float* __restrict__ Rsd,
    float* __restrict__ C, int M, int Kd, int N)
{
  __shared__ float As[2][128 * 36];   // [row][k], stride 36 fl (144B, 16B align)
  __shared__ float Bs[2][32 * 68];    // [k][n],  stride 68 fl (272B, 16B align)

  const int tid  = threadIdx.x;
  const int lane = tid & 31;
  const int wid  = tid >> 5;
  const int wm   = wid & 3;           // wave M position (0..3), 32 rows each
  const int wn   = wid >> 2;          // wave N position (0..1), 32 cols each
  const int m0   = blockIdx.y * 128;
  const int n0   = blockIdx.x * 64;

  auto stage = [&](int buf, int k0) {
    // A tile: 128x32 fp32 = 1024 16B chunks, 4 per thread
    #pragma unroll
    for (int i = 0; i < 4; i++) {
      int c = tid + i * 256;
      int row = c >> 3, col4 = (c & 7) * 4;
      cp16(&As[buf][row * 36 + col4], A + (size_t)(m0 + row) * Kd + k0 + col4);
    }
    // W tile: 32x64 fp32 = 512 16B chunks, 2 per thread
    #pragma unroll
    for (int i = 0; i < 2; i++) {
      int c = tid + i * 256;
      int row = c >> 4, col4 = (c & 15) * 4;
      cp16(&Bs[buf][row * 68 + col4], W + (size_t)(k0 + row) * N + n0 + col4);
    }
  };

  v8f acc[2][2] = {};
  const int nK = Kd >> 5;

  stage(0, 0);
  for (int kt = 0; kt < nK; kt++) {
    WAIT_ASYNC();          // my async ops for buffer kt&1 are done
    __syncthreads();       // everyone's are; previous compute finished
    if (kt + 1 < nK) stage((kt + 1) & 1, (kt + 1) << 5);   // overlap DMA

    const float* Ab = As[kt & 1];
    const float* Bb = Bs[kt & 1];

    v16bf af[2], bfg[2];
    #pragma unroll
    for (int i = 0; i < 2; i++) {
      const float* p = Ab + (wm * 32 + i * 16 + (lane & 15)) * 36;
      int kb = (lane >> 4) * 8;
      #pragma unroll
      for (int e = 0; e < 8; e++) {
        af[i][e]     = (bf16)p[kb + e];
        af[i][8 + e] = (bf16)p[16 + kb + e];
      }
    }
    #pragma unroll
    for (int j = 0; j < 2; j++) {
      int col = wn * 32 + j * 16 + (lane & 15);
      int kbb = (lane >> 4) * 8;
      #pragma unroll
      for (int e = 0; e < 8; e++) {
        bfg[j][e]     = (bf16)Bb[(kbb + e) * 68 + col];
        bfg[j][8 + e] = (bf16)Bb[(16 + kbb + e) * 68 + col];
      }
    }
    #pragma unroll
    for (int i = 0; i < 2; i++)
      #pragma unroll
      for (int j = 0; j < 2; j++)
        acc[i][j] = WMMA_BF16(af[i], bfg[j], acc[i][j]);
  }

  // epilogue: C layout VGPR r -> row (r + 8*(lane>=16)), N = lane&15
  #pragma unroll
  for (int i = 0; i < 2; i++)
    #pragma unroll
    for (int j = 0; j < 2; j++) {
      int gcol = n0 + wn * 32 + j * 16 + (lane & 15);
      float bval = bias[gcol];
      #pragma unroll
      for (int r = 0; r < 8; r++) {
        int rowL = wm * 32 + i * 16 + r + ((lane >> 4) << 3);
        size_t gi = (size_t)(m0 + rowL) * N + gcol;
        float v = acc[i][j][r] + bval;
        if (ACT) v = gelu_exact(v);
        if (RSD) v += Rsd[gi];
        C[gi] = v;
      }
    }
}

// ---------------------------------------------------------------------------
// Flash attention over the gathered subset: per (b,h), seq len K=1024.
// 4 waves/block; each wave owns a 16-row query strip of a 64-row query block.
// S = Q K^T and O += P V via WMMA; online softmax in C-layout registers.
// K/V tiles async-DMA'd raw fp32 into LDS; bf16 cvt at fragment load.
// ---------------------------------------------------------------------------
__global__ __launch_bounds__(128) void flash_attn(
    const float* __restrict__ Q, const float* __restrict__ Kb,
    const float* __restrict__ Vb, float* __restrict__ O)
{
  __shared__ float KtF[64 * 68];    // [key][dim]
  __shared__ float VtF[64 * 68];    // [key][dim]
  __shared__ bf16  Pt[4][16 * 72];  // per-wave P strip staging

  const int tid  = threadIdx.x;
  const int lane = tid & 31;
  const int w    = tid >> 5;
  const int qb   = blockIdx.x;
  const int bh   = blockIdx.y;
  const int b    = bh / Hc;
  const int h    = bh % Hc;
  const int q0   = qb * 64;
  const size_t rowbase = (size_t)b * Kc;
  const float* qp = Q  + rowbase * Dc + h * HDc;
  const float* kp = Kb + rowbase * Dc + h * HDc;
  const float* vp = Vb + rowbase * Dc + h * HDc;

  // Q strip -> two A fragments (d chunks 0-31, 32-63)
  v16bf aQ[2];
  {
    int qrow = q0 + w * 16 + (lane & 15);
    int kbb  = (lane >> 4) * 8;
    const float* src = qp + (size_t)qrow * Dc;
    #pragma unroll
    for (int c = 0; c < 2; c++)
      #pragma unroll
      for (int e = 0; e < 8; e++) {
        aQ[c][e]     = (bf16)src[c * 32 + kbb + e];
        aQ[c][8 + e] = (bf16)src[c * 32 + 16 + kbb + e];
      }
  }

  v8f Oa[4] = {};
  float mrow[8], lrow[8];
  #pragma unroll
  for (int r = 0; r < 8; r++) { mrow[r] = -3.0e38f; lrow[r] = 0.0f; }
  const float scale = 0.125f;   // 1/sqrt(64)

  for (int kblk = 0; kblk <= qb; kblk++) {
    int kbase0 = kblk * 64;
    // stage K and V blocks (64x64 fp32 each = 1024 chunks each, 8/thread)
    #pragma unroll
    for (int i = 0; i < 8; i++) {
      int c = tid + i * 128;
      int r = c >> 4, col4 = (c & 15) * 4;
      cp16(&KtF[r * 68 + col4], kp + (size_t)(kbase0 + r) * Dc + col4);
      cp16(&VtF[r * 68 + col4], vp + (size_t)(kbase0 + r) * Dc + col4);
    }
    WAIT_ASYNC();
    __syncthreads();

    // ---- S = Q K^T   (4 N-subtiles of 16, K-dim 64 = 2 wmma steps)
    v8f s[4];
    #pragma unroll
    for (int nt = 0; nt < 4; nt++) {
      v8f a0 = {};
      int key = nt * 16 + (lane & 15);
      int kbb = (lane >> 4) * 8;
      v16bf bk0, bk1;
      const float* p = KtF + key * 68;
      #pragma unroll
      for (int e = 0; e < 8; e++) {
        bk0[e] = (bf16)p[kbb + e];        // d = 0..31 chunk
        bk0[8 + e] = (bf16)p[16 + kbb + e];
        bk1[e] = (bf16)p[32 + kbb + e];   // d = 32..63 chunk
        bk1[8 + e] = (bf16)p[48 + kbb + e];
      }
      a0 = WMMA_BF16(aQ[0], bk0, a0);
      a0 = WMMA_BF16(aQ[1], bk1, a0);
      s[nt] = a0;
    }

    // ---- online softmax (rows live in 16-lane half groups)
    bool diag = (kblk == qb);
    float alpha[8];
    #pragma unroll
    for (int r = 0; r < 8; r++) {
      float mx = -3.0e38f;
      #pragma unroll
      for (int nt = 0; nt < 4; nt++) {
        float v = s[nt][r] * scale;
        if (diag) {
          int kpos = kbase0 + nt * 16 + (lane & 15);
          int qpos = q0 + w * 16 + r + ((lane >> 4) << 3);
          if (kpos > qpos) v = -1.0e9f;
        }
        s[nt][r] = v;
        mx = fmaxf(mx, v);
      }
      #pragma unroll
      for (int m = 1; m < 16; m <<= 1) mx = fmaxf(mx, __shfl_xor(mx, m, 32));
      float nm = fmaxf(mrow[r], mx);
      alpha[r] = __expf(mrow[r] - nm);
      float ps = 0.0f;
      #pragma unroll
      for (int nt = 0; nt < 4; nt++) {
        float p = __expf(s[nt][r] - nm);
        s[nt][r] = p;
        ps += p;
      }
      #pragma unroll
      for (int m = 1; m < 16; m <<= 1) ps += __shfl_xor(ps, m, 32);
      lrow[r] = lrow[r] * alpha[r] + ps;
      mrow[r] = nm;
    }

    // ---- stage P (bf16) to A-fragment-friendly layout; rescale O
    #pragma unroll
    for (int r = 0; r < 8; r++) {
      int rowM = r + ((lane >> 4) << 3);
      #pragma unroll
      for (int nt = 0; nt < 4; nt++)
        Pt[w][rowM * 72 + nt * 16 + (lane & 15)] = (bf16)s[nt][r];
      #pragma unroll
      for (int ot = 0; ot < 4; ot++) Oa[ot][r] *= alpha[r];
    }
    // per-wave Pt write->read: LDS ops from one wave stay in order; the
    // compiler inserts the dscnt wait for the dependent loads below.

    // ---- O += P V
    v16bf aP[2];
    {
      const bf16* p = &Pt[w][(lane & 15) * 72];
      int kbb = (lane >> 4) * 8;
      #pragma unroll
      for (int c = 0; c < 2; c++) {
        v8bf lo = *(const v8bf*)(p + c * 32 + kbb);
        v8bf hi = *(const v8bf*)(p + c * 32 + 16 + kbb);
        #pragma unroll
        for (int e = 0; e < 8; e++) { aP[c][e] = lo[e]; aP[c][8 + e] = hi[e]; }
      }
    }
    #pragma unroll
    for (int ot = 0; ot < 4; ot++) {
      int col = ot * 16 + (lane & 15);
      int kbb = (lane >> 4) * 8;
      v16bf bv0, bv1;
      #pragma unroll
      for (int e = 0; e < 8; e++) {
        bv0[e] = (bf16)VtF[(kbb + e) * 68 + col];        // keys 0..31
        bv0[8 + e] = (bf16)VtF[(16 + kbb + e) * 68 + col];
        bv1[e] = (bf16)VtF[(32 + kbb + e) * 68 + col];   // keys 32..63
        bv1[8 + e] = (bf16)VtF[(48 + kbb + e) * 68 + col];
      }
      Oa[ot] = WMMA_BF16(aP[0], bv0, Oa[ot]);
      Oa[ot] = WMMA_BF16(aP[1], bv1, Oa[ot]);
    }
    __syncthreads();   // protect K/V tiles before next stage overwrites
  }

  // ---- normalize and store (fp32, [B*K, D] layout, head slice)
  #pragma unroll
  for (int ot = 0; ot < 4; ot++) {
    int col = h * HDc + ot * 16 + (lane & 15);
    #pragma unroll
    for (int r = 0; r < 8; r++) {
      int rowM = r + ((lane >> 4) << 3);
      size_t orow = rowbase + q0 + w * 16 + rowM;
      O[orow * Dc + col] = Oa[ot][r] / lrow[r];
    }
  }
}

// ---------------------------------------------------------------------------
// Top-K via bitonic sort of 2048 (score, idx) pairs per batch; tie-break by
// lower index (matches jax.lax.top_k).  Then sorts kept indices ascending.
// Also accumulates the (score > 0) count for the aux loss.
// ---------------------------------------------------------------------------
__global__ __launch_bounds__(1024) void topk_sort(
    const float* __restrict__ scores, int* __restrict__ idx_out,
    float* __restrict__ fracCnt, int layer)
{
  __shared__ float sc[2048];
  __shared__ int   si[2048];
  __shared__ int   cnt;
  const int t = threadIdx.x;
  const int b = blockIdx.x;
  if (t == 0) cnt = 0;
  float s0 = scores[b * Sc + t];
  float s1 = scores[b * Sc + t + 1024];
  sc[t] = s0;        si[t] = t;
  sc[t + 1024] = s1; si[t + 1024] = t + 1024;
  int my = (s0 > 0.0f ? 1 : 0) + (s1 > 0.0f ? 1 : 0);
  __syncthreads();
  atomicAdd(&cnt, my);

  for (int kk = 2; kk <= 2048; kk <<= 1) {
    for (int j = kk >> 1; j > 0; j >>= 1) {
      __syncthreads();
      #pragma unroll
      for (int base = 0; base < 2048; base += 1024) {
        int i = base + t;
        int ixj = i ^ j;
        if (ixj > i) {
          float a = sc[i], bb = sc[ixj];
          int ia = si[i], ib = si[ixj];
          bool aFirst = (a > bb) || (a == bb && ia < ib);  // descending order
          bool up = ((i & kk) == 0);
          if (up ? !aFirst : aFirst) {
            sc[i] = bb; sc[ixj] = a; si[i] = ib; si[ixj] = ia;
          }
        }
      }
    }
  }
  __syncthreads();
  if (t == 0) atomicAdd(&fracCnt[layer], (float)cnt);

  // ascending sort of the kept K indices
  for (int kk = 2; kk <= 1024; kk <<= 1) {
    for (int j = kk >> 1; j > 0; j >>= 1) {
      __syncthreads();
      int ixj = t ^ j;
      if (ixj > t) {
        int a = si[t], bb = si[ixj];
        bool up = ((t & kk) == 0);
        if (up ? (a > bb) : (a < bb)) { si[t] = bb; si[ixj] = a; }
      }
    }
  }
  __syncthreads();
  idx_out[b * Kc + t] = si[t];
}

// ---------------------------------------------------------------------------
// LayerNorm over D=1024 per row (safe in place: inputs cached in registers).
// ---------------------------------------------------------------------------
__global__ __launch_bounds__(256) void layernorm_k(
    const float* __restrict__ X, const float* __restrict__ g,
    const float* __restrict__ bta, float* __restrict__ Y)
{
  __shared__ float red[16];
  const int row = blockIdx.x;
  const int t = threadIdx.x;
  const float* x = X + (size_t)row * Dc;
  float v[4], s = 0.0f, s2 = 0.0f;
  #pragma unroll
  for (int i = 0; i < 4; i++) { v[i] = x[t + i * 256]; s += v[i]; s2 += v[i] * v[i]; }
  #pragma unroll
  for (int m = 1; m < 32; m <<= 1) { s += __shfl_xor(s, m, 32); s2 += __shfl_xor(s2, m, 32); }
  int w = t >> 5;
  if ((t & 31) == 0) { red[w] = s; red[8 + w] = s2; }
  __syncthreads();
  float ts = 0.0f, ts2 = 0.0f;
  #pragma unroll
  for (int i = 0; i < 8; i++) { ts += red[i]; ts2 += red[8 + i]; }
  float mean = ts * (1.0f / 1024.0f);
  float var  = ts2 * (1.0f / 1024.0f) - mean * mean;
  float rstd = rsqrtf(var + 1e-5f);
  #pragma unroll
  for (int i = 0; i < 4; i++) {
    int c = t + i * 256;
    Y[(size_t)row * Dc + c] = (v[i] - mean) * rstd * g[c] + bta[c];
  }
}

// ---------------------------------------------------------------------------
// Router head: scores = rh @ Wr2 + br2  (DR=256 dot per row; 1 wave per row).
// ---------------------------------------------------------------------------
__global__ __launch_bounds__(256) void router_score_k(
    const float* __restrict__ RH, const float* __restrict__ Wr2,
    const float* __restrict__ br2, float* __restrict__ scores)
{
  int t = threadIdx.x, lane = t & 31, w = t >> 5;
  int row = blockIdx.x * 8 + w;
  const float* x = RH + (size_t)row * DRc;
  float s = 0.0f;
  #pragma unroll
  for (int i = 0; i < 8; i++) { int e = lane + i * 32; s += x[e] * Wr2[e]; }
  #pragma unroll
  for (int m = 1; m < 32; m <<= 1) s += __shfl_xor(s, m, 32);
  if (lane == 0) scores[row] = s + br2[0];
}

__global__ __launch_bounds__(256) void gather_k(
    const float* __restrict__ X, const int* __restrict__ idx, float* __restrict__ Y)
{
  int r = blockIdx.x;            // 0..B*K-1
  int b = r >> 10;               // r / K
  int src = idx[r];
  const float4* xs = (const float4*)(X + ((size_t)b * Sc + src) * Dc);
  float4* yd = (float4*)(Y + (size_t)r * Dc);
  yd[threadIdx.x] = xs[threadIdx.x];
}

__global__ __launch_bounds__(256) void scatter_k(
    const float* __restrict__ Y, const int* __restrict__ idx, float* __restrict__ X)
{
  int r = blockIdx.x;
  int b = r >> 10;
  int dst = idx[r];
  const float4* ys = (const float4*)(Y + (size_t)r * Dc);
  float4* xd = (float4*)(X + ((size_t)b * Sc + dst) * Dc);
  xd[threadIdx.x] = ys[threadIdx.x];
}

__global__ __launch_bounds__(256) void add_pos_k(
    const float* __restrict__ X, const float* __restrict__ P, float* __restrict__ Y)
{
  size_t i = ((size_t)blockIdx.x * blockDim.x + threadIdx.x) * 4;
  size_t pe = i % ((size_t)Sc * Dc);
  float4 a = *(const float4*)(X + i);
  float4 p = *(const float4*)(P + pe);
  *(float4*)(Y + i) = make_float4(a.x + p.x, a.y + p.y, a.z + p.z, a.w + p.w);
}

__global__ void zero_k(float* p, int n)
{
  int i = blockIdx.x * blockDim.x + threadIdx.x;
  if (i < n) p[i] = 0.0f;
}

__global__ void aux_k(const float* __restrict__ fracCnt, float* __restrict__ out)
{
  if (threadIdx.x == 0 && blockIdx.x == 0) {
    float aux = 0.0f, fm = 0.0f;
    for (int l = 0; l < Lc; l++) {
      float frac = fracCnt[l] / (float)(Bc * Sc);
      aux += 0.01f * (frac - 0.5f) * (frac - 0.5f);
      fm += frac;
    }
    out[0] = aux;
    out[1] = fm / (float)Lc;
  }
}

// ---------------------------------------------------------------------------
// Host orchestration.  d_out doubles as the resident x buffer [B*S, D] with
// the two scalar outputs appended.  Workspace budget ~176 MB (activations).
// ---------------------------------------------------------------------------
extern "C" void kernel_launch(void* const* d_in, const int* in_sizes, int n_in,
                              void* d_out, int out_size, void* d_ws, size_t ws_size,
                              hipStream_t stream)
{
  (void)in_sizes; (void)n_in; (void)out_size; (void)ws_size;
  const float* x_in  = (const float*)d_in[0];
  const float* pos   = (const float*)d_in[1];
  const float* Wq    = (const float*)d_in[2];
  const float* bq    = (const float*)d_in[3];
  const float* Wk    = (const float*)d_in[4];
  const float* bk    = (const float*)d_in[5];
  const float* Wv    = (const float*)d_in[6];
  const float* bv    = (const float*)d_in[7];
  const float* Wo    = (const float*)d_in[8];
  const float* bo    = (const float*)d_in[9];
  const float* gat   = (const float*)d_in[10];
  const float* bat   = (const float*)d_in[11];
  const float* gff   = (const float*)d_in[12];
  const float* bff   = (const float*)d_in[13];
  const float* W1    = (const float*)d_in[14];
  const float* b1    = (const float*)d_in[15];
  const float* W2    = (const float*)d_in[16];
  const float* b2    = (const float*)d_in[17];
  const float* Wr1   = (const float*)d_in[18];
  const float* br1   = (const float*)d_in[19];
  const float* Wr2   = (const float*)d_in[20];
  const float* br2   = (const float*)d_in[21];
  const float* gfin  = (const float*)d_in[22];
  const float* bfin  = (const float*)d_in[23];

  float* xcur = (float*)d_out;                   // [B*S, D], + 2 scalars after

  float* ws = (float*)d_ws;
  size_t off = 0;
  float* scores  = ws + off; off += (size_t)Bc * Sc;
  float* rh      = ws + off; off += (size_t)Bc * Sc * DRc;
  int*   idxb    = (int*)(ws + off); off += (size_t)Bc * Kc;
  float* xsel    = ws + off; off += (size_t)Bc * Kc * Dc;
  float* nbuf    = ws + off; off += (size_t)Bc * Kc * Dc;
  float* qbuf    = ws + off; off += (size_t)Bc * Kc * Dc;
  float* kbuf    = ws + off; off += (size_t)Bc * Kc * Dc;
  float* vbuf    = ws + off; off += (size_t)Bc * Kc * Dc;
  float* obuf    = ws + off; off += (size_t)Bc * Kc * Dc;
  float* ffh     = ws + off; off += (size_t)Bc * Kc * FFc;
  float* fracCnt = ws + off; off += 64;

  const int Mg = Bc * Kc;   // 4096 selected rows across batch

  zero_k<<<1, 64, 0, stream>>>(fracCnt, Lc);
  add_pos_k<<<(Bc * Sc * Dc / 4) / 256, 256, 0, stream>>>(x_in, pos, xcur);

  for (int l = 0; l < Lc; l++) {
    const float* Wq_l  = Wq  + (size_t)l * Dc * Dc;
    const float* bq_l  = bq  + (size_t)l * Dc;
    const float* Wk_l  = Wk  + (size_t)l * Dc * Dc;
    const float* bk_l  = bk  + (size_t)l * Dc;
    const float* Wv_l  = Wv  + (size_t)l * Dc * Dc;
    const float* bv_l  = bv  + (size_t)l * Dc;
    const float* Wo_l  = Wo  + (size_t)l * Dc * Dc;
    const float* bo_l  = bo  + (size_t)l * Dc;
    const float* ga_l  = gat + (size_t)l * Dc;
    const float* ba_l  = bat + (size_t)l * Dc;
    const float* gf_l  = gff + (size_t)l * Dc;
    const float* bf_l  = bff + (size_t)l * Dc;
    const float* W1_l  = W1  + (size_t)l * Dc * FFc;
    const float* b1_l  = b1  + (size_t)l * FFc;
    const float* W2_l  = W2  + (size_t)l * FFc * Dc;
    const float* b2_l  = b2  + (size_t)l * Dc;
    const float* Wr1_l = Wr1 + (size_t)l * Dc * DRc;
    const float* br1_l = br1 + (size_t)l * DRc;
    const float* Wr2_l = Wr2 + (size_t)l * DRc;
    const float* br2_l = br2 + (size_t)l;

    // router: h = gelu(x @ Wr1 + br1); scores = h @ Wr2 + br2
    gemm_bf16<1, 0><<<dim3(DRc / 64, (Bc * Sc) / 128), 256, 0, stream>>>(
        xcur, Wr1_l, br1_l, nullptr, rh, Bc * Sc, Dc, DRc);
    router_score_k<<<(Bc * Sc) / 8, 256, 0, stream>>>(rh, Wr2_l, br2_l, scores);
    topk_sort<<<Bc, 1024, 0, stream>>>(scores, idxb, fracCnt, l);
    gather_k<<<Mg, 256, 0, stream>>>(xcur, idxb, xsel);

    // attention block
    layernorm_k<<<Mg, 256, 0, stream>>>(xsel, ga_l, ba_l, nbuf);
    gemm_bf16<0, 0><<<dim3(Dc / 64, Mg / 128), 256, 0, stream>>>(
        nbuf, Wq_l, bq_l, nullptr, qbuf, Mg, Dc, Dc);
    gemm_bf16<0, 0><<<dim3(Dc / 64, Mg / 128), 256, 0, stream>>>(
        nbuf, Wk_l, bk_l, nullptr, kbuf, Mg, Dc, Dc);
    gemm_bf16<0, 0><<<dim3(Dc / 64, Mg / 128), 256, 0, stream>>>(
        nbuf, Wv_l, bv_l, nullptr, vbuf, Mg, Dc, Dc);
    flash_attn<<<dim3(Kc / 64, Bc * Hc), 128, 0, stream>>>(qbuf, kbuf, vbuf, obuf);
    gemm_bf16<0, 1><<<dim3(Dc / 64, Mg / 128), 256, 0, stream>>>(
        obuf, Wo_l, bo_l, xsel, xsel, Mg, Dc, Dc);

    // FFN block
    layernorm_k<<<Mg, 256, 0, stream>>>(xsel, gf_l, bf_l, nbuf);
    gemm_bf16<1, 0><<<dim3(FFc / 64, Mg / 128), 256, 0, stream>>>(
        nbuf, W1_l, b1_l, nullptr, ffh, Mg, Dc, FFc);
    gemm_bf16<0, 1><<<dim3(Dc / 64, Mg / 128), 256, 0, stream>>>(
        ffh, W2_l, b2_l, xsel, xsel, Mg, FFc, Dc);

    scatter_k<<<Mg, 256, 0, stream>>>(xsel, idxb, xcur);
  }

  layernorm_k<<<Bc * Sc, 256, 0, stream>>>(xcur, gfin, bfin, xcur);
  aux_k<<<1, 1, 0, stream>>>(fracCnt, xcur + (size_t)Bc * Sc * Dc);
}
